// TransitionDown_29085518528925
// MI455X (gfx1250) — matrix operand order repaired and checked by reading, not compiled, not run on hardware
//
#include <hip/hip_runtime.h>

typedef __attribute__((ext_vector_type(16))) _Float16 v16h;
typedef __attribute__((ext_vector_type(8)))  float    v8f;

#define B_      16
#define N_      4096
#define CIN     64
#define COUT    128
#define NPOINT_ 1024
#define KNN_    16
#define EPS_    1e-5f

// ---------------------------------------------------------------------------
// 1) Farthest point sampling: one block per batch, 1024 threads, 4 pts/thread.
//    Writes fps_idx (ws) and new_p (tail of d_out).
// ---------------------------------------------------------------------------
__global__ __launch_bounds__(1024)
void fps_kernel(const float* __restrict__ p, int* __restrict__ fps_idx,
                float* __restrict__ new_p) {
  const int b   = blockIdx.x;
  const int tid = threadIdx.x;
  const float* pb = p + (size_t)b * N_ * 3;

  float px[4], py[4], pz[4], dist[4];
#pragma unroll
  for (int t = 0; t < 4; ++t) {
    int j = tid + t * 1024;
    px[t] = pb[j * 3 + 0];
    py[t] = pb[j * 3 + 1];
    pz[t] = pb[j * 3 + 2];
    dist[t] = 1e10f;
  }

  __shared__ float s_val[32];
  __shared__ int   s_idx[32];
  __shared__ int   s_far;
  __shared__ float s_c[3];

  const int lane = tid & 31;
  const int wid  = tid >> 5;
  int farthest = 0;

  for (int i = 0; i < NPOINT_; ++i) {
    if (tid == 0) {
      fps_idx[b * NPOINT_ + i] = farthest;
      float cx = pb[farthest * 3 + 0];
      float cy = pb[farthest * 3 + 1];
      float cz = pb[farthest * 3 + 2];
      new_p[((size_t)b * NPOINT_ + i) * 3 + 0] = cx;
      new_p[((size_t)b * NPOINT_ + i) * 3 + 1] = cy;
      new_p[((size_t)b * NPOINT_ + i) * 3 + 2] = cz;
      s_c[0] = cx; s_c[1] = cy; s_c[2] = cz;
    }
    __syncthreads();
    float cx = s_c[0], cy = s_c[1], cz = s_c[2];

    float bestv = -1.0f;
    int   besti = 0x7fffffff;
#pragma unroll
    for (int t = 0; t < 4; ++t) {
      int j = tid + t * 1024;
      float dx = px[t] - cx, dy = py[t] - cy, dz = pz[t] - cz;
      float d = dx * dx + dy * dy + dz * dz;
      dist[t] = fminf(dist[t], d);
      if (dist[t] > bestv || (dist[t] == bestv && j < besti)) {
        bestv = dist[t]; besti = j;
      }
    }
    // wave32 argmax (prefer lower index on tie)
#pragma unroll
    for (int off = 16; off > 0; off >>= 1) {
      float ov = __shfl_xor(bestv, off);
      int   oi = __shfl_xor(besti, off);
      if (ov > bestv || (ov == bestv && oi < besti)) { bestv = ov; besti = oi; }
    }
    if (lane == 0) { s_val[wid] = bestv; s_idx[wid] = besti; }
    __syncthreads();
    if (tid == 0) {
      float v = s_val[0]; int ix = s_idx[0];
      for (int w = 1; w < 32; ++w)
        if (s_val[w] > v || (s_val[w] == v && s_idx[w] < ix)) { v = s_val[w]; ix = s_idx[w]; }
      s_far = ix;
    }
    __syncthreads();
    farthest = s_far;
  }
}

// ---------------------------------------------------------------------------
// 2) KNN: one block per (b, sampled point). 4096 distances in LDS, 17
//    iterative argmins (first extraction = self, dropped).
// ---------------------------------------------------------------------------
__global__ __launch_bounds__(256)
void knn_kernel(const float* __restrict__ p, const int* __restrict__ fps_idx,
                int* __restrict__ knn_idx) {
  __shared__ float sd[N_];
  __shared__ float s_v[8];
  __shared__ int   s_i[8];

  const int blk = blockIdx.x;
  const int b   = blk >> 10;          // / NPOINT_
  const int tid = threadIdx.x;
  const float* pb = p + (size_t)b * N_ * 3;

  const int ci = fps_idx[blk];
  const float cx = pb[ci * 3 + 0];
  const float cy = pb[ci * 3 + 1];
  const float cz = pb[ci * 3 + 2];

#pragma unroll 4
  for (int t = 0; t < 16; ++t) {
    int j = tid + t * 256;
    float dx = pb[j * 3 + 0] - cx;
    float dy = pb[j * 3 + 1] - cy;
    float dz = pb[j * 3 + 2] - cz;
    sd[j] = dx * dx + dy * dy + dz * dz;
  }
  __syncthreads();

  const int lane = tid & 31;
  const int wid  = tid >> 5;

  for (int it = 0; it < KNN_ + 1; ++it) {
    float bv = 3.4e38f;
    int   bi = 0x7fffffff;
#pragma unroll 4
    for (int t = 0; t < 16; ++t) {
      int j = tid + t * 256;
      float v = sd[j];
      if (v < bv || (v == bv && j < bi)) { bv = v; bi = j; }
    }
#pragma unroll
    for (int off = 16; off > 0; off >>= 1) {
      float ov = __shfl_xor(bv, off);
      int   oi = __shfl_xor(bi, off);
      if (ov < bv || (ov == bv && oi < bi)) { bv = ov; bi = oi; }
    }
    if (lane == 0) { s_v[wid] = bv; s_i[wid] = bi; }
    __syncthreads();
    if (tid == 0) {
      float v = s_v[0]; int ix = s_i[0];
      for (int w = 1; w < 8; ++w)
        if (s_v[w] < v || (s_v[w] == v && s_i[w] < ix)) { v = s_v[w]; ix = s_i[w]; }
      if (it > 0) knn_idx[(size_t)blk * KNN_ + (it - 1)] = ix;
      sd[ix] = 3.5e38f;   // exclude from next extraction
    }
    __syncthreads();
  }
}

// ---------------------------------------------------------------------------
// WMMA helpers: A-matrix 16x32 f16 fragment (diff rows), per CDNA5 ISA layout
//   lane L: row m = L%16; element e -> c = (e<8 ? e : e+8) + 8*(L>=16)
// ---------------------------------------------------------------------------
__device__ __forceinline__ void build_A(const float* __restrict__ crow,
                                        const float* __restrict__ nrow,
                                        int halfSel, v16h& a0, v16h& a1) {
#pragma unroll
  for (int e = 0; e < 16; ++e) {
    int c0 = (e < 8 ? e : e + 8) + 8 * halfSel;
    int c1 = c0 + 32;
    a0[e] = (_Float16)(crow[c0] - nrow[c0]);
    a1[e] = (_Float16)(crow[c1] - nrow[c1]);
  }
}

// ---------------------------------------------------------------------------
// 3) Pass 1: WMMA GEMM per sample (16x64 diff  x  64x128 W), accumulate
//    per-channel sum / sum-of-squares (LDS atomics -> global atomics).
// ---------------------------------------------------------------------------
__global__ __launch_bounds__(256)
void gemm_stats_kernel(const float* __restrict__ x, const float* __restrict__ conv_w,
                       const int* __restrict__ fps_idx, const int* __restrict__ knn_idx,
                       float* __restrict__ g_sum, float* __restrict__ g_sumsq) {
  __shared__ _Float16 lw[COUT * CIN];   // 16 KB weights as f16
  __shared__ float ls[COUT], lq[COUT];

  const int tid = threadIdx.x;
  for (int i = tid; i < COUT * CIN; i += 256) lw[i] = (_Float16)conv_w[i];
  if (tid < COUT) { ls[tid] = 0.0f; lq[tid] = 0.0f; }
  __syncthreads();

  const int lane = tid & 31;
  const int wave = tid >> 5;
  const int m    = lane & 15;   // neighbor (k) row
  const int hs   = lane >> 4;   // which K-half this lane supplies

  for (int s = blockIdx.x * 8 + wave; s < B_ * NPOINT_; s += gridDim.x * 8) {
    const int b = s >> 10;
    const int ci  = fps_idx[s];
    const int nbr = knn_idx[(size_t)s * KNN_ + m];
    const float* crow = x + ((size_t)b * N_ + ci)  * CIN;
    const float* nrow = x + ((size_t)b * N_ + nbr) * CIN;

    v16h a0, a1;
    build_A(crow, nrow, hs, a0, a1);

#pragma unroll
    for (int t = 0; t < 8; ++t) {
      v16h b0, b1;      // B 32x16: lane%16 = N, element = K (same order as A)
#pragma unroll
      for (int e = 0; e < 16; ++e) {
        b0[e] = lw[(16 * t + (lane & 15)) * CIN + (e < 8 ? e : e + 8) + 8 * hs];
        b1[e] = lw[(16 * t + (lane & 15)) * CIN + 32 + (e < 8 ? e : e + 8) + 8 * hs];
      }
      v8f c = {};
      c = __builtin_amdgcn_wmma_f32_16x16x32_f16(false, a0, false, b0, (short)0, c, false, false);
      c = __builtin_amdgcn_wmma_f32_16x16x32_f16(false, a1, false, b1, (short)0, c, false, false);

      float sv = 0.0f, sq = 0.0f;
#pragma unroll
      for (int r = 0; r < 8; ++r) { float v = c[r]; sv += v; sq += v * v; }
      const int o = 16 * t + (lane & 15);
      atomicAdd(&ls[o], sv);
      atomicAdd(&lq[o], sq);
    }
  }
  __syncthreads();
  if (tid < COUT) {
    atomicAdd(&g_sum[tid],   ls[tid]);
    atomicAdd(&g_sumsq[tid], lq[tid]);
  }
}

// ---------------------------------------------------------------------------
// 4) Fold stats into affine scale/bias
// ---------------------------------------------------------------------------
__global__ void finalize_stats(const float* __restrict__ g_sum, const float* __restrict__ g_sumsq,
                               const float* __restrict__ gamma, const float* __restrict__ beta,
                               float* __restrict__ scale, float* __restrict__ bias) {
  const int o = threadIdx.x;
  if (o < COUT) {
    const float M = (float)(B_ * NPOINT_ * KNN_);
    float mean = g_sum[o] / M;
    float var  = g_sumsq[o] / M - mean * mean;
    float sc   = gamma[o] * rsqrtf(var + EPS_);
    scale[o] = sc;
    bias[o]  = beta[o] - mean * sc;
  }
}

// ---------------------------------------------------------------------------
// 5) Pass 2: recompute h via WMMA, affine+ReLU, max over k, write out.
//    One wave per sample point (2048 blocks x 8 waves = 16384 samples).
// ---------------------------------------------------------------------------
__global__ __launch_bounds__(256)
void gemm_out_kernel(const float* __restrict__ x, const float* __restrict__ conv_w,
                     const int* __restrict__ fps_idx, const int* __restrict__ knn_idx,
                     const float* __restrict__ scale, const float* __restrict__ bias,
                     float* __restrict__ out) {
  __shared__ _Float16 lw[COUT * CIN];
  __shared__ float lsc[COUT], lbi[COUT];

  const int tid = threadIdx.x;
  for (int i = tid; i < COUT * CIN; i += 256) lw[i] = (_Float16)conv_w[i];
  if (tid < COUT) { lsc[tid] = scale[tid]; lbi[tid] = bias[tid]; }
  __syncthreads();

  const int lane = tid & 31;
  const int wave = tid >> 5;
  const int s = blockIdx.x * 8 + wave;
  const int b = s >> 10;
  const int m  = lane & 15;
  const int hs = lane >> 4;

  const int ci  = fps_idx[s];
  const int nbr = knn_idx[(size_t)s * KNN_ + m];
  const float* crow = x + ((size_t)b * N_ + ci)  * CIN;
  const float* nrow = x + ((size_t)b * N_ + nbr) * CIN;

  v16h a0, a1;
  build_A(crow, nrow, hs, a0, a1);

#pragma unroll
  for (int t = 0; t < 8; ++t) {
    v16h b0, b1;
#pragma unroll
    for (int e = 0; e < 16; ++e) {
      b0[e] = lw[(16 * t + (lane & 15)) * CIN + (e < 8 ? e : e + 8) + 8 * hs];
      b1[e] = lw[(16 * t + (lane & 15)) * CIN + 32 + (e < 8 ? e : e + 8) + 8 * hs];
    }
    v8f c = {};
    c = __builtin_amdgcn_wmma_f32_16x16x32_f16(false, a0, false, b0, (short)0, c, false, false);
    c = __builtin_amdgcn_wmma_f32_16x16x32_f16(false, a1, false, b1, (short)0, c, false, false);

    const int o = 16 * t + (lane & 15);
    const float sc = lsc[o], bi = lbi[o];
    float mx = 0.0f;                 // ReLU outputs are >= 0
#pragma unroll
    for (int r = 0; r < 8; ++r) {    // lane has k rows {r + 8*hs}
      float v = c[r] * sc + bi;
      v = v > 0.0f ? v : 0.0f;
      mx = v > mx ? v : mx;
    }
    float other = __shfl_xor(mx, 16);  // combine k 0-7 with k 8-15
    mx = other > mx ? other : mx;
    if (lane < 16) out[(size_t)s * COUT + o] = mx;
  }
}

// ---------------------------------------------------------------------------
__global__ void zero_stats(float* __restrict__ g_sum, float* __restrict__ g_sumsq) {
  const int t = threadIdx.x;
  if (t < COUT) { g_sum[t] = 0.0f; g_sumsq[t] = 0.0f; }
}

extern "C" void kernel_launch(void* const* d_in, const int* in_sizes, int n_in,
                              void* d_out, int out_size, void* d_ws, size_t ws_size,
                              hipStream_t stream) {
  (void)in_sizes; (void)n_in; (void)out_size; (void)ws_size;

  const float* x      = (const float*)d_in[0];   // (B, N, 64)
  const float* p      = (const float*)d_in[1];   // (B, N, 3)
  const float* conv_w = (const float*)d_in[2];   // (128, 64)
  const float* gamma  = (const float*)d_in[3];   // (128,)
  const float* beta   = (const float*)d_in[4];   // (128,)

  float* out   = (float*)d_out;                          // (B, NPOINT, 128)
  float* new_p = out + (size_t)B_ * NPOINT_ * COUT;      // (B, NPOINT, 3)

  int*   fps_idx = (int*)d_ws;                           // [B*NPOINT]
  int*   knn_idx = fps_idx + B_ * NPOINT_;               // [B*NPOINT*K]
  float* g_sum   = (float*)(knn_idx + (size_t)B_ * NPOINT_ * KNN_);
  float* g_sumsq = g_sum + COUT;
  float* scale   = g_sumsq + COUT;
  float* bias    = scale + COUT;

  zero_stats<<<1, 128, 0, stream>>>(g_sum, g_sumsq);
  fps_kernel<<<B_, 1024, 0, stream>>>(p, fps_idx, new_p);
  knn_kernel<<<B_ * NPOINT_, 256, 0, stream>>>(p, fps_idx, knn_idx);
  gemm_stats_kernel<<<256, 256, 0, stream>>>(x, conv_w, fps_idx, knn_idx, g_sum, g_sumsq);
  finalize_stats<<<1, 128, 0, stream>>>(g_sum, g_sumsq, gamma, beta, scale, bias);
  gemm_out_kernel<<<2048, 256, 0, stream>>>(x, conv_w, fps_idx, knn_idx, scale, bias, out);
}
